// RAGLiteModule_67654324847340
// MI455X (gfx1250) — matrix operand order repaired and checked by reading, not compiled, not run on hardware
//
#include <hip/hip_runtime.h>
#include <cstdint>
#include <cstddef>

#ifndef __has_builtin
#define __has_builtin(x) 0
#endif

// Async global->LDS copies (gfx1250): bypass VGPRs, tracked by ASYNCcnt.
#if defined(__AMDGCN__) && \
    __has_builtin(__builtin_amdgcn_global_load_async_to_lds_b128) && \
    __has_builtin(__builtin_amdgcn_s_wait_asynccnt)
#define RAG_ASYNC_LDS 1
#else
#define RAG_ASYNC_LDS 0
#endif

// ---------------------------------------------------------------------------
// Types / helpers
// ---------------------------------------------------------------------------
typedef __bf16 bf16_t;
typedef __attribute__((ext_vector_type(16))) __bf16 v16bf;
typedef __attribute__((ext_vector_type(8)))  float  v8f;
typedef __attribute__((ext_vector_type(4)))  int    v4i;

__device__ __forceinline__ bf16_t f2bf(float x) {
  union { float f; uint32_t u; } v; v.f = x;
  uint32_t r = v.u + 0x7FFFu + ((v.u >> 16) & 1u);   // round-to-nearest-even
  union { uint16_t s; bf16_t h; } o; o.s = (uint16_t)(r >> 16);
  return o.h;
}

__device__ __forceinline__ float gelu_exact(float x) {
  return 0.5f * x * (1.0f + erff(x * 0.7071067811865475f));
}

union FragBF16 { uint4 u[2]; v16bf v; };

#if RAG_ASYNC_LDS
// 16B async copy: global -> LDS, no VGPR round-trip (global_load_async_to_lds_b128)
// Builtin signature: (v4i addrspace(1)*, v4i addrspace(3)*, imm offset, imm cpol).
// Flat shared pointers carry the LDS byte offset in addr[31:0] (ISA 10.2),
// so the inttoptr round-trip yields the correct AS(3) pointer.
__device__ __forceinline__ void async_copy16(const void* gsrc, void* lds_dst) {
  __attribute__((address_space(1))) v4i* g =
      (__attribute__((address_space(1))) v4i*)(uintptr_t)gsrc;
  __attribute__((address_space(3))) v4i* l =
      (__attribute__((address_space(3))) v4i*)(uint32_t)(uintptr_t)lds_dst;
  __builtin_amdgcn_global_load_async_to_lds_b128(g, l, 0, 0);
}
#endif

// ---------------------------------------------------------------------------
// fp32 -> bf16 cast (grid-stride)
// ---------------------------------------------------------------------------
__global__ __launch_bounds__(256)
void k_cast_bf16(const float* __restrict__ s, bf16_t* __restrict__ d, size_t n) {
  size_t i = (size_t)blockIdx.x * blockDim.x + threadIdx.x;
  size_t stride = (size_t)gridDim.x * blockDim.x;
  for (; i < n; i += stride) d[i] = f2bf(s[i]);
}

// ---------------------------------------------------------------------------
// Tiled bf16 GEMM:  C[M,N] = A[M,K] * B[N,K]^T (+bias[n]) (+act)
// block = 256 thr = 8 waves; block tile 64(M) x 128(N); BK = 32
// Double-buffered LDS, async global->LDS staging, 2x2 WMMA per wave.
// ACT: 0 = none, 1 = exact gelu. Writes f32 and/or bf16 outputs.
// ---------------------------------------------------------------------------
template<int ACT>
__global__ __launch_bounds__(256, 2)
void k_gemm_bf16(const bf16_t* __restrict__ A, const bf16_t* __restrict__ B,
                 const float* __restrict__ bias,
                 float* __restrict__ outF, bf16_t* __restrict__ outH,
                 int M, int N, int K) {
  __shared__ __align__(16) bf16_t As[2][64][32];    // 2 x 4 KB
  __shared__ __align__(16) bf16_t Bs[2][128][32];   // 2 x 8 KB

  const int tid  = threadIdx.x;
  const int lane = tid & 31;
  const int wid  = tid >> 5;       // 0..7
  const int wm   = wid >> 2;       // 0..1 (M)
  const int wn   = wid & 3;        // 0..3 (N)
  const int lm   = lane & 15;      // row/col within 16
  const int lh   = lane >> 4;      // half-wave selects K-range

  const int m0 = blockIdx.y * 64;
  const int n0 = blockIdx.x * 128;

  v8f c[2][2];
  const v8f vzero = {0.f,0.f,0.f,0.f,0.f,0.f,0.f,0.f};
#pragma unroll
  for (int mi = 0; mi < 2; ++mi)
#pragma unroll
    for (int ni = 0; ni < 2; ++ni) c[mi][ni] = vzero;

  const int arow  = tid >> 2;        // 0..63
  const int acol8 = (tid & 3) * 8;   // bf16 element offset (16B groups)
  const int nsteps = K / 32;

  // Stage a 32-deep K tile into LDS buffer `buf`
  auto issue_stage = [&](int step, int buf) {
    const int k0 = step * 32;
    const bf16_t* pa  = A + (size_t)(m0 + arow) * K + k0 + acol8;
    const bf16_t* pb0 = B + (size_t)(n0 + arow) * K + k0 + acol8;
    const bf16_t* pb1 = B + (size_t)(n0 + arow + 64) * K + k0 + acol8;
#if RAG_ASYNC_LDS
    async_copy16(pa,  &As[buf][arow][acol8]);
    async_copy16(pb0, &Bs[buf][arow][acol8]);
    async_copy16(pb1, &Bs[buf][arow + 64][acol8]);
#else
    const uint4 ta  = *reinterpret_cast<const uint4*>(pa);
    const uint4 tb0 = *reinterpret_cast<const uint4*>(pb0);
    const uint4 tb1 = *reinterpret_cast<const uint4*>(pb1);
    *reinterpret_cast<uint4*>(&As[buf][arow][acol8])      = ta;
    *reinterpret_cast<uint4*>(&Bs[buf][arow][acol8])      = tb0;
    *reinterpret_cast<uint4*>(&Bs[buf][arow + 64][acol8]) = tb1;
    if (step + 1 < nsteps) {   // hint next tile toward the caches
      __builtin_prefetch(pa + 32, 0, 1);
      __builtin_prefetch(pb0 + 32, 0, 1);
    }
#endif
  };

  issue_stage(0, 0);

  for (int step = 0; step < nsteps; ++step) {
    const int buf = step & 1;
    const bool more = (step + 1) < nsteps;
    if (more) issue_stage(step + 1, buf ^ 1);   // overlap next copy with math

#if RAG_ASYNC_LDS
    // 3 async ops per wave per stage, in-order completion:
    // <=3 outstanding  =>  the older stage has fully landed in LDS.
    if (more) __builtin_amdgcn_s_wait_asynccnt(3);
    else      __builtin_amdgcn_s_wait_asynccnt(0);
#endif
    __syncthreads();

    // A fragment (16x32): lanes 0-15 -> K{0..7,16..23}, lanes 16-31 -> K{8..15,24..31}
    FragBF16 a[2], b[2];
#pragma unroll
    for (int mi = 0; mi < 2; ++mi) {
      const char* base = (const char*)&As[buf][wm * 32 + mi * 16 + lm][0];
      a[mi].u[0] = *(const uint4*)(base + lh * 16);
      a[mi].u[1] = *(const uint4*)(base + 32 + lh * 16);
    }
    // B fragment (32x16): lanes 0-15 -> K{0..7,16..23} of col n, lanes 16-31 -> K{8..15,24..31}
#pragma unroll
    for (int ni = 0; ni < 2; ++ni) {
      const char* base = (const char*)&Bs[buf][wn * 32 + ni * 16 + lm][0];
      b[ni].u[0] = *(const uint4*)(base + lh * 16);
      b[ni].u[1] = *(const uint4*)(base + 32 + lh * 16);
    }
#pragma unroll
    for (int mi = 0; mi < 2; ++mi)
#pragma unroll
      for (int ni = 0; ni < 2; ++ni)
        c[mi][ni] = __builtin_amdgcn_wmma_f32_16x16x32_bf16(
            false, a[mi].v, false, b[ni].v, (short)0, c[mi][ni], false, false);
    __syncthreads();   // protect buffer reuse (WAR for stage step+2)
  }

  // Epilogue: C/D layout -> lane m = r + lh*8, n = lm
#pragma unroll
  for (int mi = 0; mi < 2; ++mi) {
#pragma unroll
    for (int ni = 0; ni < 2; ++ni) {
      const int mbase = m0 + wm * 32 + mi * 16 + lh * 8;
      const int n     = n0 + wn * 32 + ni * 16 + lm;
      const float bv  = bias ? bias[n] : 0.0f;
#pragma unroll
      for (int r = 0; r < 8; ++r) {
        float v = c[mi][ni][r] + bv;
        if (ACT == 1) v = gelu_exact(v);
        const size_t off = (size_t)(mbase + r) * N + n;
        if (outF) outF[off] = v;
        if (outH) outH[off] = f2bf(v);
      }
    }
  }
}

// ---------------------------------------------------------------------------
// LayerNorm(E=128) + L2 normalize, one wave per token -> bf16 qe
// ---------------------------------------------------------------------------
__global__ __launch_bounds__(256)
void k_ln_l2(const float* __restrict__ h2, const float* __restrict__ g,
             const float* __restrict__ b, bf16_t* __restrict__ qe) {
  const int lane  = threadIdx.x & 31;
  const int wid   = threadIdx.x >> 5;
  const int token = blockIdx.x * 8 + wid;

  const float4 xv = *(const float4*)(h2 + (size_t)token * 128 + lane * 4);
  float x[4] = {xv.x, xv.y, xv.z, xv.w};

  float s = x[0] + x[1] + x[2] + x[3];
#pragma unroll
  for (int o = 16; o > 0; o >>= 1) s += __shfl_xor(s, o);
  const float mu = s * (1.0f / 128.0f);

  float vs = 0.f;
#pragma unroll
  for (int i = 0; i < 4; ++i) { float d = x[i] - mu; vs += d * d; }
#pragma unroll
  for (int o = 16; o > 0; o >>= 1) vs += __shfl_xor(vs, o);
  const float rstd = rsqrtf(vs * (1.0f / 128.0f) + 1e-5f);

  const float4 gv = *(const float4*)(g + lane * 4);
  const float4 bv = *(const float4*)(b + lane * 4);
  const float ga[4] = {gv.x, gv.y, gv.z, gv.w};
  const float ba[4] = {bv.x, bv.y, bv.z, bv.w};

  float y[4], n2 = 0.f;
#pragma unroll
  for (int i = 0; i < 4; ++i) { y[i] = (x[i] - mu) * rstd * ga[i] + ba[i]; n2 += y[i] * y[i]; }
#pragma unroll
  for (int o = 16; o > 0; o >>= 1) n2 += __shfl_xor(n2, o);
  const float inv = 1.0f / fmaxf(sqrtf(n2), 1e-12f);

#pragma unroll
  for (int i = 0; i < 4; ++i)
    qe[(size_t)token * 128 + lane * 4 + i] = f2bf(y[i] * inv);
}

// ---------------------------------------------------------------------------
// Top-3 over N=1024 + softmax + position gather, one wave per token
// ---------------------------------------------------------------------------
__device__ __forceinline__ void top3_insert(float v, int i,
    float& s0, float& s1, float& s2, int& i0, int& i1, int& i2) {
  if (v > s0 || (v == s0 && i < i0))      { s2=s1; i2=i1; s1=s0; i1=i0; s0=v; i0=i; }
  else if (v > s1 || (v == s1 && i < i1)) { s2=s1; i2=i1; s1=v;  i1=i; }
  else if (v > s2 || (v == s2 && i < i2)) { s2=v;  i2=i; }
}

__global__ __launch_bounds__(256)
void k_topk(const float* __restrict__ sim, const int* __restrict__ ipos,
            float* __restrict__ topw, int* __restrict__ toppos) {
  const int lane  = threadIdx.x & 31;
  const int wid   = threadIdx.x >> 5;
  const int token = blockIdx.x * 8 + wid;
  const float* row = sim + (size_t)token * 1024;

  float s0 = -3.4e38f, s1 = -3.4e38f, s2 = -3.4e38f;
  int   i0 = 0x7fffffff, i1 = 0x7fffffff, i2 = 0x7fffffff;
  for (int n = lane; n < 1024; n += 32)
    top3_insert(row[n], n, s0, s1, s2, i0, i1, i2);

#pragma unroll
  for (int o = 16; o > 0; o >>= 1) {
    float t0 = __shfl_xor(s0, o), t1 = __shfl_xor(s1, o), t2 = __shfl_xor(s2, o);
    int   j0 = __shfl_xor(i0, o), j1 = __shfl_xor(i1, o), j2 = __shfl_xor(i2, o);
    top3_insert(t0, j0, s0, s1, s2, i0, i1, i2);
    top3_insert(t1, j1, s0, s1, s2, i0, i1, i2);
    top3_insert(t2, j2, s0, s1, s2, i0, i1, i2);
  }

  if (lane == 0) {
    const float e0 = 1.0f, e1 = expf(s1 - s0), e2 = expf(s2 - s0);
    const float inv = 1.0f / (e0 + e1 + e2);
    const int idx[3] = {i0, i1, i2};
    const float w[3] = {e0 * inv, e1 * inv, e2 * inv};
#pragma unroll
    for (int k = 0; k < 3; ++k) {
      int p = ipos[idx[k]];
      p = p < 0 ? 0 : (p > 1023 ? 1023 : p);
      toppos[token * 3 + k] = p;
      topw[token * 3 + k]   = w[k];
    }
  }
}

// ---------------------------------------------------------------------------
// context = sum_k w_k * values[pos_k]; write ctx f32 + cat bf16 [hs | ctx]
// one block per token, H = 2048
// ---------------------------------------------------------------------------
__global__ __launch_bounds__(256)
void k_ctx_cat(const float* __restrict__ hs, const float* __restrict__ values,
               const float* __restrict__ topw, const int* __restrict__ toppos,
               float* __restrict__ ctx, bf16_t* __restrict__ cat, int H) {
  const int token = blockIdx.x;
  const float w0 = topw[token * 3 + 0], w1 = topw[token * 3 + 1], w2 = topw[token * 3 + 2];
  const float* v0 = values + (size_t)toppos[token * 3 + 0] * H;
  const float* v1 = values + (size_t)toppos[token * 3 + 1] * H;
  const float* v2 = values + (size_t)toppos[token * 3 + 2] * H;
  const float* hrow = hs + (size_t)token * H;
  bf16_t* crow = cat + (size_t)token * 2 * H;
  float*  xrow = ctx + (size_t)token * H;

  for (int e = threadIdx.x; e < H; e += 256) {
    const float cv = w0 * v0[e] + w1 * v1[e] + w2 * v2[e];
    xrow[e]     = cv;
    crow[e]     = f2bf(hrow[e]);
    crow[H + e] = f2bf(cv);
  }
}

// ---------------------------------------------------------------------------
// gate = sigmoid(dot(cat, Wg) + bg), one block per token (2H = 4096)
// ---------------------------------------------------------------------------
__global__ __launch_bounds__(256)
void k_gate(const float* __restrict__ hs, const float* __restrict__ ctx,
            const float* __restrict__ Wg, const float* __restrict__ bg,
            float* __restrict__ gate, int H) {
  __shared__ float red[256];
  const int token = blockIdx.x;
  float acc = 0.f;
  for (int ccc = threadIdx.x; ccc < 2 * H; ccc += 256) {
    const float x = (ccc < H) ? hs[(size_t)token * H + ccc]
                              : ctx[(size_t)token * H + (ccc - H)];
    acc += x * Wg[ccc];
  }
  red[threadIdx.x] = acc;
  __syncthreads();
  for (int s = 128; s > 0; s >>= 1) {
    if (threadIdx.x < s) red[threadIdx.x] += red[threadIdx.x + s];
    __syncthreads();
  }
  if (threadIdx.x == 0)
    gate[token] = 1.0f / (1.0f + expf(-(red[0] + bg[0])));
}

// ---------------------------------------------------------------------------
// out = hs + gate[token] * fused2
// ---------------------------------------------------------------------------
__global__ __launch_bounds__(256)
void k_final(const float* __restrict__ hs, const float* __restrict__ f2,
             const float* __restrict__ gate, float* __restrict__ out,
             int H, size_t n) {
  size_t i = (size_t)blockIdx.x * blockDim.x + threadIdx.x;
  size_t stride = (size_t)gridDim.x * blockDim.x;
  for (; i < n; i += stride)
    out[i] = hs[i] + gate[i / H] * f2[i];
}

// ---------------------------------------------------------------------------
// Host launcher
// ---------------------------------------------------------------------------
extern "C" void kernel_launch(void* const* d_in, const int* in_sizes, int n_in,
                              void* d_out, int out_size, void* d_ws, size_t ws_size,
                              hipStream_t stream) {
  (void)in_sizes; (void)n_in; (void)out_size; (void)ws_size;

  const float* hs   = (const float*)d_in[0];
  const float* idxe = (const float*)d_in[1];
  const int*   ipos = (const int*)  d_in[2];
  const float* vals = (const float*)d_in[3];
  const float* Wq   = (const float*)d_in[4];
  const float* bq   = (const float*)d_in[5];
  const float* W1   = (const float*)d_in[6];
  const float* b1   = (const float*)d_in[7];
  const float* W2   = (const float*)d_in[8];
  const float* b2   = (const float*)d_in[9];
  const float* lng  = (const float*)d_in[10];
  const float* lnb  = (const float*)d_in[11];
  const float* Wg   = (const float*)d_in[12];
  const float* bg   = (const float*)d_in[13];
  const float* Wf1  = (const float*)d_in[14];
  const float* bf1  = (const float*)d_in[15];
  const float* Wf2  = (const float*)d_in[16];
  const float* bf2  = (const float*)d_in[17];
  float* out = (float*)d_out;

  constexpr int Hc = 2048, Ec = 128, Nc = 1024, T = 4 * 2048; // 8192 tokens
  constexpr int Hh = Hc / 2;                                   // 1024

  char* ws = (char*)d_ws;
  size_t off = 0;
  auto alloc = [&](size_t bytes) -> char* {
    char* p = ws + off;
    off += (bytes + 255) & ~(size_t)255;
    return p;
  };

  bf16_t* hsB  = (bf16_t*)alloc((size_t)T * Hc * 2);
  bf16_t* WqB  = (bf16_t*)alloc((size_t)Hc * Hc * 2);
  bf16_t* W1B  = (bf16_t*)alloc((size_t)Hh * Hc * 2);
  bf16_t* W2B  = (bf16_t*)alloc((size_t)Ec * Hh * 2);
  bf16_t* Wf1B = (bf16_t*)alloc((size_t)Hc * 2 * Hc * 2);
  bf16_t* Wf2B = (bf16_t*)alloc((size_t)Hc * Hc * 2);
  bf16_t* idxB = (bf16_t*)alloc((size_t)Nc * Ec * 2);
  bf16_t* qB   = (bf16_t*)alloc((size_t)T * Hc * 2);
  bf16_t* h1B  = (bf16_t*)alloc((size_t)T * Hh * 2);
  float*  h2F  = (float*) alloc((size_t)T * Ec * 4);
  bf16_t* qeB  = (bf16_t*)alloc((size_t)T * Ec * 2);
  float*  simF = (float*) alloc((size_t)T * Nc * 4);
  float*  topw = (float*) alloc((size_t)T * 3 * 4);
  int*    tpos = (int*)   alloc((size_t)T * 3 * 4);
  float*  ctxF = (float*) alloc((size_t)T * Hc * 4);
  bf16_t* catB = (bf16_t*)alloc((size_t)T * 2 * Hc * 2);
  bf16_t* f1B  = (bf16_t*)alloc((size_t)T * Hc * 2);
  float*  f2F  = (float*) alloc((size_t)T * Hc * 4);
  float*  gatF = (float*) alloc((size_t)T * 4);

  auto cast = [&](const float* s, bf16_t* d, size_t n) {
    k_cast_bf16<<<4096, 256, 0, stream>>>(s, d, n);
  };
  cast(hs,   hsB,  (size_t)T * Hc);
  cast(Wq,   WqB,  (size_t)Hc * Hc);
  cast(W1,   W1B,  (size_t)Hh * Hc);
  cast(W2,   W2B,  (size_t)Ec * Hh);
  cast(Wf1,  Wf1B, (size_t)Hc * 2 * Hc);
  cast(Wf2,  Wf2B, (size_t)Hc * Hc);
  cast(idxe, idxB, (size_t)Nc * Ec);

  // 1) q = hs @ Wq^T + bq                       [8192 x 2048], K=2048
  k_gemm_bf16<0><<<dim3(Hc / 128, T / 64), 256, 0, stream>>>(
      hsB, WqB, bq, nullptr, qB, T, Hc, Hc);
  // 2) h1 = gelu(q @ W1^T + b1)                 [8192 x 1024], K=2048
  k_gemm_bf16<1><<<dim3(Hh / 128, T / 64), 256, 0, stream>>>(
      qB, W1B, b1, nullptr, h1B, T, Hh, Hc);
  // 3) h2 = h1 @ W2^T + b2                      [8192 x 128], K=1024
  k_gemm_bf16<0><<<dim3(Ec / 128, T / 64), 256, 0, stream>>>(
      h1B, W2B, b2, h2F, nullptr, T, Ec, Hh);
  // 4) layernorm + L2 -> qe (bf16)
  k_ln_l2<<<T / 8, 256, 0, stream>>>(h2F, lng, lnb, qeB);
  // 5) sim = qe @ index^T                       [8192 x 1024], K=128
  k_gemm_bf16<0><<<dim3(Nc / 128, T / 64), 256, 0, stream>>>(
      qeB, idxB, nullptr, simF, nullptr, T, Nc, Ec);
  // 6) top-3 + softmax + gather positions
  k_topk<<<T / 8, 256, 0, stream>>>(simF, ipos, topw, tpos);
  // 7) context + concat (bf16)
  k_ctx_cat<<<T, 256, 0, stream>>>(hs, vals, topw, tpos, ctxF, catB, Hc);
  // 8) gate
  k_gate<<<T, 256, 0, stream>>>(hs, ctxF, Wg, bg, gatF, Hc);
  // 9) f1 = gelu(cat @ Wf1^T + bf1)             [8192 x 2048], K=4096
  k_gemm_bf16<1><<<dim3(Hc / 128, T / 64), 256, 0, stream>>>(
      catB, Wf1B, bf1, nullptr, f1B, T, Hc, 2 * Hc);
  // 10) f2 = f1 @ Wf2^T + bf2                   [8192 x 2048], K=2048
  k_gemm_bf16<0><<<dim3(Hc / 128, T / 64), 256, 0, stream>>>(
      f1B, Wf2B, bf2, f2F, nullptr, T, Hc, Hc);
  // 11) out = hs + gate * f2
  k_final<<<8192, 256, 0, stream>>>(hs, f2F, gatF, out, Hc, (size_t)T * Hc);
}
